// TrafficXLSTM_49168785605065
// MI455X (gfx1250) — matrix operand order repaired and checked by reading, not compiled, not run on hardware
//
#include <hip/hip_runtime.h>
#include <cstdint>
#include <cstddef>

// ---------------- model dims ----------------
#define B_     64
#define S_     256
#define D_     512
#define INNER_ 1024
#define NHM    4
#define DHM    256     // INNER/NHM
#define NHS    4
#define DHS    128     // D/NHS
#define FFUP   640
#define R_     (B_*S_) // 16384 rows
#define EPS_   1e-5f

typedef __attribute__((ext_vector_type(16))) _Float16 v16h;
typedef __attribute__((ext_vector_type(8)))  _Float16 v8h;
typedef __attribute__((ext_vector_type(8)))  float    v8f;

__device__ __forceinline__ float siluf(float x) { return x / (1.f + expf(-x)); }

// =====================================================================
// Weight convert + transpose: W[K][N] f32 -> Wt[N][K] f16 (LDS-tiled)
// grid (N/32, K/32), 256 threads
// =====================================================================
__global__ __launch_bounds__(256) void k_w2h(
    const float* __restrict__ W, int K, int N, _Float16* __restrict__ Wt)
{
  __shared__ float t[32][33];
  int n0 = blockIdx.x * 32, k0 = blockIdx.y * 32;
  int tx = threadIdx.x & 31, ty = threadIdx.x >> 5;  // ty 0..7
  #pragma unroll
  for (int j = 0; j < 4; ++j) {
    int k = k0 + ty + j * 8;
    t[ty + j * 8][tx] = W[(size_t)k * N + n0 + tx];
  }
  __syncthreads();
  #pragma unroll
  for (int j = 0; j < 4; ++j) {
    int n = n0 + ty + j * 8;
    Wt[(size_t)n * K + k0 + tx] = (_Float16)t[tx][ty + j * 8];
  }
}

// =====================================================================
// WMMA GEMM: C[M,N] = act(A[M,K] @ W[K,N] + bias) (+ resid)
// A: f16 row-major (lda). Wt: f16 transposed [N][K] (ldk == K).
// M%128==0, N%128==0, K%64==0 at all call sites.
// act: 0=none, 1=silu, 2=gelu(exact)
// Register double-buffer: next tile loads overlap current tile WMMAs.
// =====================================================================
#define TM 128
#define TN 128
#define TK 64

__global__ __launch_bounds__(256) void gemm_wmma(
    const _Float16* __restrict__ A, int lda,
    const _Float16* __restrict__ Wt, int ldk,
    const float* __restrict__ bias,
    const float* __restrict__ resid, int ldr,
    float* __restrict__ C, int ldc,
    int M, int N, int K, int act)
{
  __shared__ _Float16 As[TM * TK];   // [m][k]  16KB
  __shared__ _Float16 Ws[TN * TK];   // [n][k]  16KB

  const int tid  = threadIdx.x;
  const int lane = tid & 31;
  const int wave = tid >> 5;
  const int wm   = wave & 3;    // 4 row groups of 32
  const int wn   = wave >> 2;   // 2 col groups of 64
  const int m0   = blockIdx.x * TM;
  const int n0   = blockIdx.y * TN;
  const int r    = tid >> 1;          // staging row (A) / col (W), 0..127
  const int cb   = (tid & 1) * 32;    // 32-half chunk

  const _Float16* Ap = A  + (size_t)(m0 + r) * lda + cb;
  const _Float16* Wp = Wt + (size_t)(n0 + r) * ldk + cb;

  v8f acc[2][4] = {};
  v8h ra[4], rw[4];
  #pragma unroll
  for (int i = 0; i < 4; ++i) { ra[i] = *(const v8h*)(Ap + i * 8); rw[i] = *(const v8h*)(Wp + i * 8); }

  const int koff = (lane & 16) ? 8 : 0;   // A frag: lanes>=16 carry K+8
  const int kf   = (lane & 16) ? 8 : 0;   // B frag: lanes>=16 carry K+8
  const int ksteps = K / TK;

  for (int ks = 0; ks < ksteps; ++ks) {
    __syncthreads();
    #pragma unroll
    for (int i = 0; i < 4; ++i) {
      *(v8h*)&As[r * TK + cb + i * 8] = ra[i];
      *(v8h*)&Ws[r * TK + cb + i * 8] = rw[i];
    }
    __syncthreads();
    if (ks + 1 < ksteps) {
      #pragma unroll
      for (int i = 0; i < 4; ++i) {
        ra[i] = *(const v8h*)(Ap + (ks + 1) * TK + i * 8);
        rw[i] = *(const v8h*)(Wp + (ks + 1) * TK + i * 8);
      }
    }
    #pragma unroll
    for (int kk = 0; kk < 2; ++kk) {
      v16h af[2], bf[4];
      #pragma unroll
      for (int mt = 0; mt < 2; ++mt) {
        int row = wm * 32 + mt * 16 + (lane & 15);
        v8h lo = *(const v8h*)&As[row * TK + kk * 32 + koff];
        v8h hi = *(const v8h*)&As[row * TK + kk * 32 + 16 + koff];
        #pragma unroll
        for (int i = 0; i < 8; ++i) { af[mt][i] = lo[i]; af[mt][8 + i] = hi[i]; }
      }
      #pragma unroll
      for (int nt = 0; nt < 4; ++nt) {
        int col = wn * 64 + nt * 16 + (lane & 15);
        v8h lo = *(const v8h*)&Ws[col * TK + kk * 32 + kf];
        v8h hi = *(const v8h*)&Ws[col * TK + kk * 32 + kf + 16];
        #pragma unroll
        for (int i = 0; i < 8; ++i) { bf[nt][i] = lo[i]; bf[nt][8 + i] = hi[i]; }
      }
      #pragma unroll
      for (int mt = 0; mt < 2; ++mt)
        #pragma unroll
        for (int nt = 0; nt < 4; ++nt)
          acc[mt][nt] = __builtin_amdgcn_wmma_f32_16x16x32_f16(
              false, af[mt], false, bf[nt], (short)0, acc[mt][nt], false, false);
    }
  }

  // epilogue: C f32 layout (vgpr r -> M=r, lanes>=16 -> M=r+8)
  #pragma unroll
  for (int mt = 0; mt < 2; ++mt) {
    #pragma unroll
    for (int nt = 0; nt < 4; ++nt) {
      int row = m0 + wm * 32 + mt * 16 + ((lane & 16) ? 8 : 0);
      int col = n0 + wn * 64 + nt * 16 + (lane & 15);
      float bv = bias ? bias[col] : 0.f;
      #pragma unroll
      for (int rr = 0; rr < 8; ++rr) {
        float v = acc[mt][nt][rr] + bv;
        if (act == 1)      v = siluf(v);
        else if (act == 2) v = 0.5f * v * (1.f + erff(v * 0.70710678f));
        if (resid) v += resid[(size_t)(row + rr) * ldr + col];
        C[(size_t)(row + rr) * ldc + col] = v;
      }
    }
  }
}

// =====================================================================
// Elementwise / small kernels
// =====================================================================
__global__ void k_embed(const float* __restrict__ X, const float* __restrict__ Wm,
                        const float* __restrict__ Bv, float* __restrict__ H, int total)
{
  int idx = blockIdx.x * 256 + threadIdx.x;
  if (idx >= total) return;
  int d = idx % D_; size_t bs = (size_t)idx / D_;
  float acc = Bv[d];
  #pragma unroll
  for (int i = 0; i < 3; ++i) acc += X[bs * 3 + i] * Wm[i * D_ + d];
  H[bs * D_ + d] = acc;
}

// layernorm (width 512); optional f16 copy for GEMM consumers
__global__ __launch_bounds__(128) void k_layernorm(
    const float* __restrict__ X, const float* __restrict__ W,
    float* __restrict__ Y, _Float16* __restrict__ Yh)
{
  size_t row = blockIdx.x;
  int tid = threadIdx.x, lane = tid & 31, wave = tid >> 5;
  const float* x = X + row * D_;
  float v[4], s = 0.f, ss = 0.f;
  #pragma unroll
  for (int i = 0; i < 4; ++i) { v[i] = x[tid * 4 + i]; s += v[i]; ss += v[i] * v[i]; }
  #pragma unroll
  for (int off = 16; off; off >>= 1) { s += __shfl_xor(s, off, 32); ss += __shfl_xor(ss, off, 32); }
  __shared__ float s1[4], s2[4];
  if (lane == 0) { s1[wave] = s; s2[wave] = ss; }
  __syncthreads();
  float ts  = s1[0] + s1[1] + s1[2] + s1[3];
  float tss = s2[0] + s2[1] + s2[2] + s2[3];
  float mu = ts / D_, var = tss / D_ - mu * mu;
  float rs = rsqrtf(var + EPS_);
  #pragma unroll
  for (int i = 0; i < 4; ++i) {
    int j = tid * 4 + i;
    float y = (v[i] - mu) * rs * W[j];
    Y[row * D_ + j] = y;
    if (Yh) Yh[row * D_ + j] = (_Float16)y;
  }
}

// causal conv1d (K=4) + SiLU; either/both f32 and f16 outputs
__global__ void k_conv(const float* __restrict__ X, int xstride,
                       const float* __restrict__ W, const float* __restrict__ Bv,
                       float* __restrict__ YF, _Float16* __restrict__ YH,
                       int C, int total)
{
  int idx = blockIdx.x * 256 + threadIdx.x;
  if (idx >= total) return;
  int c = idx % C; size_t bs = (size_t)idx / C; int s = (int)(bs % S_);
  float acc = Bv[c];
  #pragma unroll
  for (int k = 0; k < 4; ++k) {
    int sp = s - 3 + k;
    if (sp >= 0) acc += X[(bs - (size_t)(3 - k)) * xstride + c] * W[k * C + c];
  }
  acc = siluf(acc);
  if (YF) YF[bs * (size_t)C + c] = acc;
  if (YH) YH[bs * (size_t)C + c] = (_Float16)acc;
}

// head-wise 4x4 q/k/v projections; f32 for gates + f16 (q pre-scaled) for WMMA
__global__ void k_qkv(const float* __restrict__ XC, const float* __restrict__ U,
                      const float* __restrict__ qw, const float* __restrict__ qb,
                      const float* __restrict__ kw, const float* __restrict__ kb,
                      const float* __restrict__ vw, const float* __restrict__ vb,
                      float* __restrict__ Q, float* __restrict__ K, float* __restrict__ V,
                      _Float16* __restrict__ Qh, _Float16* __restrict__ Kh)
{
  int idx = blockIdx.x * 256 + threadIdx.x;  // (row, group)
  int g = idx & 255; size_t row = (size_t)idx >> 8;
  float xc[4], xm[4];
  #pragma unroll
  for (int j = 0; j < 4; ++j) {
    xc[j] = XC[row * INNER_ + g * 4 + j];
    xm[j] = U[row * 2048 + g * 4 + j];
  }
  #pragma unroll
  for (int i = 0; i < 4; ++i) {
    float aq = qb[g * 4 + i], ak = kb[g * 4 + i], av = vb[g * 4 + i];
    #pragma unroll
    for (int j = 0; j < 4; ++j) {
      aq += xc[j] * qw[g * 16 + j * 4 + i];
      ak += xc[j] * kw[g * 16 + j * 4 + i];
      av += xm[j] * vw[g * 16 + j * 4 + i];
    }
    size_t o = row * INNER_ + g * 4 + i;
    Q[o] = aq; K[o] = ak; V[o] = av;
    Qh[o] = (_Float16)(aq * 0.0625f);   // fold dh^-1/2
    Kh[o] = (_Float16)ak;
  }
}

// V f32 (B,S,INNER) -> Vt f16 [(b*4+h)][d][s]  (LDS-tiled transpose)
__global__ __launch_bounds__(256) void k_vtrans(
    const float* __restrict__ V, _Float16* __restrict__ Vt)
{
  __shared__ _Float16 t[32][33];
  int bh = blockIdx.z; int b = bh >> 2, hh = bh & 3;
  int s0 = blockIdx.x * 32, d0 = blockIdx.y * 32;
  int tx = threadIdx.x & 31, ty = threadIdx.x >> 5;
  #pragma unroll
  for (int j = 0; j < 4; ++j) {
    int s = s0 + ty + j * 8;
    t[ty + j * 8][tx] = (_Float16)V[((size_t)(b * S_ + s)) * INNER_ + hh * DHM + d0 + tx];
  }
  __syncthreads();
  #pragma unroll
  for (int j = 0; j < 4; ++j) {
    int d = d0 + ty + j * 8;
    Vt[((size_t)bh * DHM + d) * S_ + s0 + tx] = t[tx][ty + j * 8];
  }
}

// ig/fg = concat(q,k,v) @ W(3072x4) + b   -> stored (B, NH, S)
__global__ __launch_bounds__(256) void k_gates(
    const float* __restrict__ Q, const float* __restrict__ K, const float* __restrict__ V,
    const float* __restrict__ igw, const float* __restrict__ igb,
    const float* __restrict__ fgw, const float* __restrict__ fgb,
    float* __restrict__ IG, float* __restrict__ FG)
{
  size_t row = blockIdx.x;
  int tid = threadIdx.x, lane = tid & 31, wave = tid >> 5;
  int b = (int)(row / S_), s = (int)(row % S_);
  float pi[4] = {0,0,0,0}, pf[4] = {0,0,0,0};
  for (int j = tid; j < 3 * INNER_; j += 256) {
    float xv = (j < INNER_)     ? Q[row * INNER_ + j]
             : (j < 2 * INNER_) ? K[row * INNER_ + (j - INNER_)]
                                : V[row * INNER_ + (j - 2 * INNER_)];
    #pragma unroll
    for (int h = 0; h < 4; ++h) { pi[h] += xv * igw[(size_t)j * 4 + h]; pf[h] += xv * fgw[(size_t)j * 4 + h]; }
  }
  #pragma unroll
  for (int h = 0; h < 4; ++h)
    #pragma unroll
    for (int off = 16; off; off >>= 1) { pi[h] += __shfl_xor(pi[h], off, 32); pf[h] += __shfl_xor(pf[h], off, 32); }
  __shared__ float red[8][8];
  if (lane == 0) {
    #pragma unroll
    for (int h = 0; h < 4; ++h) { red[wave][h] = pi[h]; red[wave][h + 4] = pf[h]; }
  }
  __syncthreads();
  if (tid < 8) {
    float sum = 0.f;
    #pragma unroll
    for (int w = 0; w < 8; ++w) sum += red[w][tid];
    int hh = tid & 3;
    if (tid < 4) IG[((size_t)b * NHM + hh) * S_ + s] = sum + igb[hh];
    else         FG[((size_t)b * NHM + hh) * S_ + s] = sum + fgb[hh];
  }
}

// =====================================================================
// mLSTM parallel attention: one workgroup per (b, head).
// LDS: aux + Q (256x256 f16) + score matrix C (256x256 f16) ~260KB.
// Causal structure exploited: upper-triangle tiles never computed
// (csh pre-zeroed with b128 stores); pass-2 K-range clamped per wave.
// Q staged via GLOBAL_LOAD_ASYNC_TO_LDS (ASYNCcnt path).
// =====================================================================
__global__ __launch_bounds__(256) void k_attn(
    const _Float16* __restrict__ Qh, const _Float16* __restrict__ Kh,
    const _Float16* __restrict__ Vt,
    const float* __restrict__ IG, const float* __restrict__ FG,
    float* __restrict__ HA)
{
  extern __shared__ char smem[];
  float* lfc = (float*)smem;                 // S_+4
  float* igs = lfc + (S_ + 4);
  float* mxd = igs + S_;
  float* nrm = mxd + S_;
  _Float16* qsh = (_Float16*)(nrm + S_);     // S_*DHM
  _Float16* csh = qsh + (size_t)S_ * DHM;    // S_*S_

  const int tid  = threadIdx.x;
  const int lane = tid & 31;
  const int wave = tid >> 5;
  const int b  = blockIdx.x >> 2;
  const int hh = blockIdx.x & 3;
  const int s  = tid;

  // ---- async-stage Q row into LDS (global -> LDS, no VGPR round-trip) ----
  {
    const _Float16* src = Qh + ((size_t)(b * S_ + s)) * INNER_ + hh * DHM;
    unsigned lds = (unsigned)(size_t)(qsh + (size_t)s * DHM);
    #pragma unroll
    for (int i = 0; i < DHM / 8; ++i) {
      asm volatile("global_load_async_to_lds_b128 %0, %1, off"
                   :: "v"(lds + i * 16), "v"(src + i * 8)
                   : "memory");
    }
  }
  // ---- zero the score matrix (enables triangular skipping below) ----
  {
    v8h z = {};
    v8h* cz = (v8h*)csh;
    #pragma unroll 4
    for (int i = tid; i < (S_ * S_) / 8; i += 256) cz[i] = z;
  }
  float fgv = FG[((size_t)b * NHM + hh) * S_ + s];
  lfc[s + 1] = fminf(fgv, 0.f) - log1pf(expf(-fabsf(fgv)));   // log_sigmoid
  igs[s] = IG[((size_t)b * NHM + hh) * S_ + s];
  asm volatile("s_wait_asynccnt 0" ::: "memory");
  __syncthreads();
  if (tid == 0) { lfc[0] = 0.f; for (int t = 0; t < S_; ++t) lfc[t + 1] += lfc[t]; }
  __syncthreads();
  {
    float mm = -3.0e38f;
    for (int t = 0; t <= s; ++t) mm = fmaxf(mm, lfc[s + 1] - lfc[t + 1] + igs[t]);
    mxd[s] = mm;
  }
  __syncthreads();

  const int rbase = wave * 32;
  const int koff  = (lane & 16) ? 8 : 0;
  const int kf    = (lane & 16) ? 8 : 0;

  // ---------- pass 1: C = (q kT) * exp(logD - maxD), rowsums ----------
  // rows of this wave: [rbase, rbase+31] -> only column tiles ct <= 2*wave+1
  for (int mt = 0; mt < 2; ++mt) {
    float rsum[8];
    #pragma unroll
    for (int r = 0; r < 8; ++r) rsum[r] = 0.f;
    const int arow = rbase + mt * 16 + (lane & 15);
    const int ctmax = 2 * wave + 1;
    for (int ct = 0; ct <= ctmax; ++ct) {
      v8f acc = {};
      const int tcol = ct * 16 + (lane & 15);
      const _Float16* kbase = Kh + ((size_t)(b * S_ + tcol)) * INNER_ + hh * DHM;
      for (int ks = 0; ks < DHM / 32; ++ks) {
        v16h af, bf;
        v8h alo = *(const v8h*)&qsh[(size_t)arow * DHM + ks * 32 + koff];
        v8h ahi = *(const v8h*)&qsh[(size_t)arow * DHM + ks * 32 + 16 + koff];
        v8h blo = *(const v8h*)(kbase + ks * 32 + kf);
        v8h bhi = *(const v8h*)(kbase + ks * 32 + kf + 16);
        #pragma unroll
        for (int i = 0; i < 8; ++i) { af[i] = alo[i]; af[8 + i] = ahi[i]; bf[i] = blo[i]; bf[8 + i] = bhi[i]; }
        acc = __builtin_amdgcn_wmma_f32_16x16x32_f16(false, af, false, bf, (short)0, acc, false, false);
      }
      #pragma unroll
      for (int r = 0; r < 8; ++r) {
        int row = rbase + mt * 16 + r + ((lane & 16) ? 8 : 0);
        float val = 0.f;
        if (tcol <= row) {
          float ld = lfc[row + 1] - lfc[tcol + 1] + igs[tcol];
          val = acc[r] * __expf(ld - mxd[row]);
        }
        csh[(size_t)row * S_ + tcol] = (_Float16)val;
        rsum[r] += val;
      }
    }
    #pragma unroll
    for (int r = 0; r < 8; ++r) {
      rsum[r] += __shfl_xor(rsum[r], 1, 16);
      rsum[r] += __shfl_xor(rsum[r], 2, 16);
      rsum[r] += __shfl_xor(rsum[r], 4, 16);
      rsum[r] += __shfl_xor(rsum[r], 8, 16);
    }
    if ((lane & 15) == 0) {
      int row0 = rbase + mt * 16 + ((lane & 16) ? 8 : 0);
      #pragma unroll
      for (int r = 0; r < 8; ++r) nrm[row0 + r] = rsum[r];
    }
  }
  __syncthreads();
  nrm[s] = 1.f / (fmaxf(fabsf(nrm[s]), __expf(-mxd[s])) + 1e-6f);
  __syncthreads();

  // ---------- pass 2: out = (C / norm) @ V ----------
  // C[row][t] == 0 for t > rbase+31  ->  K loop clamped to ks <= wave
  for (int nt = 0; nt < DHM / 16; ++nt) {
    for (int mt = 0; mt < 2; ++mt) {
      v8f acc = {};
      const int arow = rbase + mt * 16 + (lane & 15);
      const _Float16 sc = (_Float16)nrm[arow];
      const int dcol = nt * 16 + (lane & 15);
      const _Float16* vbase = Vt + ((size_t)blockIdx.x * DHM + dcol) * S_;
      for (int ks = 0; ks <= wave; ++ks) {
        v16h af, bf;
        v8h alo = *(const v8h*)&csh[(size_t)arow * S_ + ks * 32 + koff];
        v8h ahi = *(const v8h*)&csh[(size_t)arow * S_ + ks * 32 + 16 + koff];
        v8h blo = *(const v8h*)(vbase + ks * 32 + kf);
        v8h bhi = *(const v8h*)(vbase + ks * 32 + kf + 16);
        #pragma unroll
        for (int i = 0; i < 8; ++i) { af[i] = alo[i] * sc; af[8 + i] = ahi[i] * sc; bf[i] = blo[i]; bf[8 + i] = bhi[i]; }
        acc = __builtin_amdgcn_wmma_f32_16x16x32_f16(false, af, false, bf, (short)0, acc, false, false);
      }
      #pragma unroll
      for (int r = 0; r < 8; ++r) {
        int row = rbase + mt * 16 + r + ((lane & 16) ? 8 : 0);
        HA[((size_t)(b * S_ + row)) * INNER_ + hh * DHM + nt * 16 + (lane & 15)] = acc[r];
      }
    }
  }
}

// head-norm + skip*xc, then * silu(z) -> f16 (only consumed by down GEMM)
__global__ __launch_bounds__(256) void k_mix(
    const float* __restrict__ HA, const float* __restrict__ XC,
    const float* __restrict__ U, const float* __restrict__ MHW,
    const float* __restrict__ SK, _Float16* __restrict__ OUTH)
{
  size_t row = blockIdx.x;
  int tid = threadIdx.x, lane = tid & 31, wave = tid >> 5;
  const float* h = HA + row * INNER_;
  float v[4], s = 0.f, ss = 0.f;
  #pragma unroll
  for (int i = 0; i < 4; ++i) { v[i] = h[tid * 4 + i]; s += v[i]; ss += v[i] * v[i]; }
  #pragma unroll
  for (int off = 16; off; off >>= 1) { s += __shfl_xor(s, off, 32); ss += __shfl_xor(ss, off, 32); }
  __shared__ float s1[8], s2[8];
  if (lane == 0) { s1[wave] = s; s2[wave] = ss; }
  __syncthreads();
  int head = tid >> 6;
  float ts  = s1[2 * head] + s1[2 * head + 1];
  float tss = s2[2 * head] + s2[2 * head + 1];
  float mu = ts / DHM, var = tss / DHM - mu * mu;
  float rs = rsqrtf(var + EPS_);
  const float* z = U + row * 2048 + INNER_;
  #pragma unroll
  for (int i = 0; i < 4; ++i) {
    int j = tid * 4 + i;
    float hn = (v[i] - mu) * rs * MHW[j] + SK[j] * XC[row * INNER_ + j];
    OUTH[row * INNER_ + j] = (_Float16)(hn * siluf(z[j]));
  }
}

// =====================================================================
// sLSTM sequential scan: 256 independent (b, head) chains.
// =====================================================================
__global__ __launch_bounds__(128) void k_scan(
    const float* __restrict__ PRE,  // (B,S,4,NHS,DHS) row stride 2048
    const float* __restrict__ Rw,   // (4,NHS,DHS,DHS)
    float* __restrict__ HS)         // (B,S,D)
{
  int b  = blockIdx.x >> 2;
  int nh = blockIdx.x & 3;
  int e  = threadIdx.x;
  __shared__ float hsh[DHS];
  float c = 0.f, n = 0.f, m = 0.f;
  hsh[e] = 0.f;
  __syncthreads();
  for (int t = 0; t < S_; ++t) {
    float g[4];
    #pragma unroll
    for (int gi = 0; gi < 4; ++gi) {
      float acc = PRE[((size_t)(b * S_ + t)) * 2048 + gi * 512 + nh * DHS + e];
      const float* Rp = Rw + ((size_t)(gi * NHS + nh)) * DHS * DHS + e;
      for (int d = 0; d < DHS; ++d) acc += hsh[d] * Rp[(size_t)d * DHS];
      g[gi] = acc;
    }
    float m2 = fmaxf(g[1] + m, g[0]);
    float i  = __expf(g[0] - m2);
    float f  = __expf(g[1] + m - m2);
    float c2 = f * c + i * tanhf(g[2]);
    float n2 = f * n + i;
    float h2 = (1.f / (1.f + __expf(-g[3]))) * c2 / n2;
    c = c2; n = n2; m = m2;
    HS[((size_t)(b * S_ + t)) * D_ + nh * DHS + e] = h2;
    __syncthreads();
    hsh[e] = h2;
    __syncthreads();
  }
}

// sLSTM head-norm with residual add into the stream
__global__ __launch_bounds__(128) void k_slstm_norm(
    const float* __restrict__ H, const float* __restrict__ W, float* __restrict__ X)
{
  size_t row = blockIdx.x;
  int tid = threadIdx.x;
  const float* h = H + row * D_;
  float v[4], s = 0.f, ss = 0.f;
  #pragma unroll
  for (int i = 0; i < 4; ++i) { v[i] = h[tid * 4 + i]; s += v[i]; ss += v[i] * v[i]; }
  #pragma unroll
  for (int off = 16; off; off >>= 1) { s += __shfl_xor(s, off, 32); ss += __shfl_xor(ss, off, 32); }
  float mu = s / DHS, var = ss / DHS - mu * mu;
  float rs = rsqrtf(var + EPS_);
  #pragma unroll
  for (int i = 0; i < 4; ++i) { int j = tid * 4 + i; X[row * D_ + j] += (v[i] - mu) * rs * W[j]; }
}

// gelu(gate)*val -> f16 (only consumed by FFN down GEMM)
__global__ void k_geglu(const float* __restrict__ U, _Float16* __restrict__ YH, int total)
{
  int idx = blockIdx.x * 256 + threadIdx.x;
  if (idx >= total) return;
  int j = idx % FFUP; size_t row = (size_t)idx / FFUP;
  float g = U[row * (2 * FFUP) + j], val = U[row * (2 * FFUP) + FFUP + j];
  YH[row * FFUP + j] = (_Float16)(0.5f * g * (1.f + erff(g * 0.70710678f)) * val);
}

__global__ void k_outproj(const float* __restrict__ Hc, const float* __restrict__ W,
                          const float* __restrict__ Bv, float* __restrict__ OUT)
{
  int idx = threadIdx.x;
  if (idx >= B_ * 3) return;
  int b = idx / 3, o = idx % 3;
  const float* h = Hc + ((size_t)(b * S_ + S_ - 1)) * D_;
  float acc = Bv[o];
  for (int d = 0; d < D_; ++d) acc += h[d] * W[d * 3 + o];
  OUT[b * 3 + o] = acc;
}

// =====================================================================
// Host orchestration
// =====================================================================
extern "C" void kernel_launch(void* const* d_in, const int* in_sizes, int n_in,
                              void* d_out, int out_size, void* d_ws, size_t ws_size,
                              hipStream_t stream)
{
  (void)in_sizes; (void)n_in; (void)out_size; (void)ws_size;
  const float* x       = (const float*)d_in[0];
  const float* emb_w   = (const float*)d_in[1];
  const float* emb_b   = (const float*)d_in[2];
  const float* out_w   = (const float*)d_in[3];
  const float* out_b   = (const float*)d_in[4];
  const float* post_ln = (const float*)d_in[5];

  const size_t R = R_;
  float* ws  = (float*)d_ws;
  float* bLN = ws;                 // R x 512
  float* bU  = bLN + R * 512;      // R x 2048 (up / pre / ffn-up)
  float* bXC = bU  + R * 2048;     // R x 1024 (mLSTM conv out f32)
  float* bQ  = bXC + R * 1024;     // R x 1024 (gates input)
  float* bK  = bQ  + R * 1024;
  float* bV  = bK  + R * 1024;     // also sLSTM hs
  float* bHA = bV  + R * 1024;
  float* bH  = bHA + R * 1024;     // residual stream
  float* bIG = bH  + R * 512;
  float* bFG = bIG + (size_t)B_ * NHM * S_;
  _Float16* hp   = (_Float16*)(bFG + (size_t)B_ * NHM * S_);
  _Float16* hLN  = hp;             // R x 512
  _Float16* hXC  = hLN + R * 512;  // R x 1024 (sLSTM conv f16 / GEGLU f16)
  _Float16* hMIX = hXC + R * 1024; // R x 1024
  _Float16* hQ   = hMIX + R * 1024;// R x 1024 (scaled)
  _Float16* hK   = hQ  + R * 1024;
  _Float16* hVt  = hK  + R * 1024; // (B*4, 256, 256)
  _Float16* hWt  = hVt + R * 1024; // weight scratch (max 512*2048 = 1M halves)

  const size_t attn_smem = (size_t)(S_ + 4 + 3 * S_) * sizeof(float)
                         + (size_t)S_ * DHM * sizeof(_Float16)
                         + (size_t)S_ * S_ * sizeof(_Float16);

  k_embed<<<(unsigned)((R * D_ + 255) / 256), 256, 0, stream>>>(x, emb_w, emb_b, bH, (int)(R * D_));

  int idx = 6;
  for (int blk = 0; blk < 6; ++blk) {
    if (blk == 1) {
      // ---------------- sLSTM block ----------------
      const float* ln_w   = (const float*)d_in[idx + 0];
      const float* conv_w = (const float*)d_in[idx + 1];
      const float* conv_b = (const float*)d_in[idx + 2];
      const float* gate_w = (const float*)d_in[idx + 3];
      const float* rec_w  = (const float*)d_in[idx + 4];
      const float* gate_b = (const float*)d_in[idx + 5];
      const float* gn_w   = (const float*)d_in[idx + 6];
      const float* ln2_w  = (const float*)d_in[idx + 7];
      const float* fup_w  = (const float*)d_in[idx + 8];
      const float* fup_b  = (const float*)d_in[idx + 9];
      const float* fdn_w  = (const float*)d_in[idx + 10];
      const float* fdn_b  = (const float*)d_in[idx + 11];
      idx += 12;

      k_layernorm<<<(unsigned)R, 128, 0, stream>>>(bH, ln_w, bLN, hLN);
      k_conv<<<(unsigned)((R * 512 + 255) / 256), 256, 0, stream>>>(bLN, 512, conv_w, conv_b, nullptr, hXC, 512, (int)(R * 512));
      for (int gn = 0; gn < 16; ++gn)
        k_w2h<<<dim3(4, 4), 256, 0, stream>>>(gate_w + (size_t)gn * 128 * 128, 128, 128, hWt + (size_t)gn * 128 * 128);
      for (int g = 0; g < 4; ++g)
        for (int nhd = 0; nhd < 4; ++nhd) {
          const _Float16* Ain = (g < 2 ? hXC : hLN) + nhd * 128;
          gemm_wmma<<<dim3((unsigned)(R / 128), 1), 256, 0, stream>>>(
              Ain, 512,
              hWt + (size_t)(g * 4 + nhd) * 128 * 128, 128,
              gate_b + (size_t)(g * 4 + nhd) * 128,
              nullptr, 0,
              bU + g * 512 + nhd * 128, 2048,
              (int)R, 128, 128, 0);
        }
      k_scan<<<B_ * NHS, 128, 0, stream>>>(bU, rec_w, bV);
      k_slstm_norm<<<(unsigned)R, 128, 0, stream>>>(bV, gn_w, bH);
      k_layernorm<<<(unsigned)R, 128, 0, stream>>>(bH, ln2_w, bLN, hLN);
      k_w2h<<<dim3((2 * FFUP) / 32, 512 / 32), 256, 0, stream>>>(fup_w, 512, 2 * FFUP, hWt);
      gemm_wmma<<<dim3((unsigned)(R / 128), (2 * FFUP) / 128), 256, 0, stream>>>(
          hLN, 512, hWt, 512, fup_b, nullptr, 0, bU, 2 * FFUP, (int)R, 2 * FFUP, 512, 0);
      k_geglu<<<(unsigned)((R * FFUP + 255) / 256), 256, 0, stream>>>(bU, hXC, (int)(R * FFUP));
      k_w2h<<<dim3(D_ / 32, FFUP / 32), 256, 0, stream>>>(fdn_w, FFUP, D_, hWt);
      gemm_wmma<<<dim3((unsigned)(R / 128), D_ / 128), 256, 0, stream>>>(
          hXC, FFUP, hWt, FFUP, fdn_b, bH, D_, bH, D_, (int)R, D_, FFUP, 0);
    } else {
      // ---------------- mLSTM block ----------------
      const float* ln_w   = (const float*)d_in[idx + 0];
      const float* up_w   = (const float*)d_in[idx + 1];
      const float* up_b   = (const float*)d_in[idx + 2];
      const float* conv_w = (const float*)d_in[idx + 3];
      const float* conv_b = (const float*)d_in[idx + 4];
      const float* q_w    = (const float*)d_in[idx + 5];
      const float* q_b    = (const float*)d_in[idx + 6];
      const float* k_w    = (const float*)d_in[idx + 7];
      const float* k_b    = (const float*)d_in[idx + 8];
      const float* v_w    = (const float*)d_in[idx + 9];
      const float* v_b    = (const float*)d_in[idx + 10];
      const float* ig_w   = (const float*)d_in[idx + 11];
      const float* ig_b   = (const float*)d_in[idx + 12];
      const float* fg_w   = (const float*)d_in[idx + 13];
      const float* fg_b   = (const float*)d_in[idx + 14];
      const float* mh_w   = (const float*)d_in[idx + 15];
      const float* skip_w = (const float*)d_in[idx + 16];
      const float* down_w = (const float*)d_in[idx + 17];
      const float* down_b = (const float*)d_in[idx + 18];
      idx += 19;

      k_layernorm<<<(unsigned)R, 128, 0, stream>>>(bH, ln_w, bLN, hLN);
      k_w2h<<<dim3((2 * INNER_) / 32, D_ / 32), 256, 0, stream>>>(up_w, 512, 2 * INNER_, hWt);
      gemm_wmma<<<dim3((unsigned)(R / 128), (2 * INNER_) / 128), 256, 0, stream>>>(
          hLN, 512, hWt, 512, up_b, nullptr, 0, bU, 2 * INNER_, (int)R, 2 * INNER_, 512, 0);
      k_conv<<<(unsigned)((R * INNER_ + 255) / 256), 256, 0, stream>>>(bU, 2048, conv_w, conv_b, bXC, nullptr, INNER_, (int)(R * INNER_));
      k_qkv<<<(unsigned)R, 256, 0, stream>>>(bXC, bU, q_w, q_b, k_w, k_b, v_w, v_b, bQ, bK, bV, hQ, hK);
      k_gates<<<(unsigned)R, 256, 0, stream>>>(bQ, bK, bV, ig_w, ig_b, fg_w, fg_b, bIG, bFG);
      k_vtrans<<<dim3(8, 8, B_ * NHM), 256, 0, stream>>>(bV, hVt);
      k_attn<<<B_ * NHM, 256, attn_smem, stream>>>(hQ, hK, hVt, bIG, bFG, bHA);
      k_mix<<<(unsigned)R, 256, 0, stream>>>(bHA, bXC, bU, mh_w, skip_w, hMIX);
      k_w2h<<<dim3(D_ / 32, INNER_ / 32), 256, 0, stream>>>(down_w, INNER_, D_, hWt);
      gemm_wmma<<<dim3((unsigned)(R / 128), D_ / 128), 256, 0, stream>>>(
          hMIX, INNER_, hWt, INNER_, down_b, bH, D_, bH, D_, (int)R, D_, INNER_, 0);
    }
  }

  k_layernorm<<<(unsigned)R, 128, 0, stream>>>(bH, post_ln, bLN, nullptr);
  k_outproj<<<1, 256, 0, stream>>>(bLN, out_w, out_b, (float*)d_out);
}
